// ThingFusion_49289044688904
// MI455X (gfx1250) — compile-verified, matrix operand order checked by reading, not run on hardware
//
#include <hip/hip_runtime.h>

// ThingFusion on MI455X (gfx1250):
//  K1: row-normalize (f32 math) -> bf16 matrix in ws, minHit[j]=j
//  K2: upper-triangular X·X^T via v_wmma_f32_16x16x32_bf16, 32x32 tile per
//      wave (2x2 register blocking -> 2 b128 loads per WMMA). K loop kept
//      ROLLED (unroll 1) so the live set fits in VGPRs -> no scratch spills.
//  K3: keep/cats/scores outputs, zero fused + counts
//  K4: sparse scatter-add: column j adds k[j] into row firstHit[j] (if kept)
//  K5: divide kept rows by their count
//
// N=8192, D=256 fixed by the reference.

#define NN 8192
#define DD 256
#define THRESH 0.95f

typedef __attribute__((ext_vector_type(16))) __bf16        v16bf;
typedef __attribute__((ext_vector_type(8)))  float         v8f;
typedef __attribute__((ext_vector_type(4)))  float         v4f;
typedef __attribute__((ext_vector_type(4)))  unsigned int  v4u;

union Frag {
    v4u   u[2];   // 32 bytes = 16 bf16
    v16bf v;
};

__device__ __forceinline__ unsigned short f32_to_bf16(float f) {
    unsigned int u = __float_as_uint(f);
    u += 0x7FFFu + ((u >> 16) & 1u);   // round-to-nearest-even
    return (unsigned short)(u >> 16);
}

// ---------------- K1: normalize rows to bf16, init minHit ----------------
// 1 wave per row, 8 rows per 256-thread block. Lane owns 8 contiguous elems.
__global__ void k_normalize(const float* __restrict__ kern,
                            unsigned short* __restrict__ abf,
                            int* __restrict__ minHit) {
    const int lane = threadIdx.x & 31;
    const int row  = (blockIdx.x << 3) + (threadIdx.x >> 5);

    const float* src = kern + (size_t)row * DD + lane * 8;
    v4f x0 = *(const v4f*)(src);
    v4f x1 = *(const v4f*)(src + 4);

    float ss = x0.x*x0.x + x0.y*x0.y + x0.z*x0.z + x0.w*x0.w
             + x1.x*x1.x + x1.y*x1.y + x1.z*x1.z + x1.w*x1.w;
    #pragma unroll
    for (int off = 16; off > 0; off >>= 1)
        ss += __shfl_xor(ss, off, 32);

    const float inv = 1.0f / fmaxf(sqrtf(ss), 1e-8f);

    unsigned short h[8];
    #pragma unroll
    for (int t = 0; t < 4; ++t) h[t]     = f32_to_bf16(x0[t] * inv);
    #pragma unroll
    for (int t = 0; t < 4; ++t) h[4 + t] = f32_to_bf16(x1[t] * inv);

    v4u p;
    p.x = (unsigned)h[0] | ((unsigned)h[1] << 16);
    p.y = (unsigned)h[2] | ((unsigned)h[3] << 16);
    p.z = (unsigned)h[4] | ((unsigned)h[5] << 16);
    p.w = (unsigned)h[6] | ((unsigned)h[7] << 16);
    *(v4u*)(abf + (size_t)row * DD + lane * 8) = p;

    if (lane == 0) minHit[row] = row;   // diagonal always labels itself
}

// ---------------- K2: WMMA similarity, 32x32 tile / wave ------------------
// Block = 256 threads = 8 waves arranged 2(row) x 4(col); block tile 64x128.
// For X·X^T the A (16x32, lane=M, K packed per lane) and B (32x16, lane=N,
// K packed per lane) fragments load rows of the bf16 matrix with the same
// per-lane layout. Only two base addresses; +16-row / +K16 fragments are
// reached via the 24-bit instruction immediate offset.
__device__ __forceinline__ void tile_epilogue(const v8f& acc, int rb, int cb,
                                              const int* __restrict__ cats,
                                              int* __restrict__ minHit,
                                              int lane) {
    const int j  = cb + (lane & 15);
    const int cj = cats[j];
    #pragma unroll
    for (int r = 0; r < 8; ++r) {
        const int i = rb + r + ((lane >> 4) << 3);   // C layout: M=r+8*(l>>4)
        if (i <= j && acc[r] >= THRESH && cats[i] == cj)
            atomicMin(&minHit[j], i);
    }
}

__global__ void k_sim(const unsigned short* __restrict__ abf,
                      const int* __restrict__ cats,
                      int* __restrict__ minHit) {
    const int wave = threadIdx.x >> 5;
    const int lane = threadIdx.x & 31;

    const int rowBase = blockIdx.y * 64  + ((wave >> 2) << 5);  // 32-row strip
    const int colBase = blockIdx.x * 128 + ((wave & 3) << 5);   // 32-col strip
    if (colBase + 31 < rowBase) return;   // wave-uniform: tile fully below diag

    const int mrow  = lane & 15;          // fragment row/col within 16-tile
    const int khalf = (lane >> 4) << 3;   // lanes 16-31 hold the K+8 half

    // v4u == 16 bytes == 8 bf16. One 16-row step = 16*DD/8 = 512 v4u = 8192 B.
    const v4u* aP = (const v4u*)(abf + (size_t)(rowBase + mrow) * DD + khalf);
    const v4u* bP = (const v4u*)(abf + (size_t)(colBase + mrow) * DD + khalf);

    v8f acc00 = {0.f,0.f,0.f,0.f,0.f,0.f,0.f,0.f};
    v8f acc01 = acc00, acc10 = acc00, acc11 = acc00;

    #pragma unroll 1
    for (int kk = 0; kk < DD; kk += 32, aP += 4, bP += 4) {
        Frag a0, a1, b0, b1;
        a0.u[0] = aP[0];    a0.u[1] = aP[2];     // rows rowBase..,   K halves
        a1.u[0] = aP[512];  a1.u[1] = aP[514];   // rows rowBase+16..
        b0.u[0] = bP[0];    b0.u[1] = bP[2];     // cols colBase..
        b1.u[0] = bP[512];  b1.u[1] = bP[514];   // cols colBase+16..
        acc00 = __builtin_amdgcn_wmma_f32_16x16x32_bf16(
                    false, a0.v, false, b0.v, (short)0, acc00, false, false);
        acc01 = __builtin_amdgcn_wmma_f32_16x16x32_bf16(
                    false, a0.v, false, b1.v, (short)0, acc01, false, false);
        acc10 = __builtin_amdgcn_wmma_f32_16x16x32_bf16(
                    false, a1.v, false, b0.v, (short)0, acc10, false, false);
        acc11 = __builtin_amdgcn_wmma_f32_16x16x32_bf16(
                    false, a1.v, false, b1.v, (short)0, acc11, false, false);
    }

    tile_epilogue(acc00, rowBase,      colBase,      cats, minHit, lane);
    tile_epilogue(acc01, rowBase,      colBase + 16, cats, minHit, lane);
    tile_epilogue(acc10, rowBase + 16, colBase,      cats, minHit, lane);
    tile_epilogue(acc11, rowBase + 16, colBase + 16, cats, minHit, lane);
}

// ---------------- K3: scalar outputs + zero fused/cnt ----------------
__global__ void k_prep(const int* __restrict__ cats,
                       const float* __restrict__ scores,
                       const int* __restrict__ minHit,
                       float* __restrict__ fused,
                       float* __restrict__ cats_out,
                       float* __restrict__ scores_out,
                       float* __restrict__ keep_out,
                       int* __restrict__ cnt) {
    const int lane = threadIdx.x & 31;
    const int row  = (blockIdx.x << 3) + (threadIdx.x >> 5);

    const v4f z = {0.f, 0.f, 0.f, 0.f};
    float* dst = fused + (size_t)row * DD + lane * 8;
    *(v4f*)(dst)     = z;
    *(v4f*)(dst + 4) = z;

    if (lane == 0) {
        const bool kp   = (minHit[row] == row);
        cats_out[row]   = kp ? (float)cats[row] : -1.0f;
        scores_out[row] = kp ? scores[row] : 0.0f;
        keep_out[row]   = kp ? 1.0f : 0.0f;
        cnt[row]        = 0;
    }
}

// ---------------- K4: sparse scatter-add (<=1 nonzero per column) ----------
__global__ void k_aggregate(const float* __restrict__ kern,
                            const int* __restrict__ minHit,
                            int* __restrict__ cnt,
                            float* __restrict__ fused) {
    const int lane = threadIdx.x & 31;
    const int col  = (blockIdx.x << 3) + (threadIdx.x >> 5);

    const int i = minHit[col];          // first row to hit column `col`
    if (minHit[i] != i) return;         // that row was itself dropped

    if (lane == 0) atomicAdd(&cnt[i], 1);

    const float* src = kern + (size_t)col * DD;
    float*       dst = fused + (size_t)i * DD;
    #pragma unroll
    for (int t = 0; t < 8; ++t) {
        const int d = lane + (t << 5);
        atomicAdd(&dst[d], src[d]);
    }
}

// ---------------- K5: divide kept rows by their member count --------------
__global__ void k_finalize(const int* __restrict__ minHit,
                           const int* __restrict__ cnt,
                           float* __restrict__ fused) {
    const int lane = threadIdx.x & 31;
    const int row  = (blockIdx.x << 3) + (threadIdx.x >> 5);
    if (minHit[row] != row) return;     // dropped rows stay zero

    const int c = cnt[row];
    const float inv = 1.0f / (float)(c > 0 ? c : 1);
    float* dst = fused + (size_t)row * DD;
    #pragma unroll
    for (int t = 0; t < 8; ++t)
        dst[lane + (t << 5)] *= inv;
}

extern "C" void kernel_launch(void* const* d_in, const int* in_sizes, int n_in,
                              void* d_out, int out_size, void* d_ws, size_t ws_size,
                              hipStream_t stream) {
    (void)in_sizes; (void)n_in; (void)out_size; (void)ws_size;

    const float* kern   = (const float*)d_in[0];   // [1,N,D] f32
    const int*   cats   = (const int*)d_in[1];     // [1,N]   int
    const float* scores = (const float*)d_in[2];   // [1,N]   f32

    float* out        = (float*)d_out;             // fused | cats | scores | keep
    float* fused      = out;
    float* cats_out   = out + (size_t)NN * DD;
    float* scores_out = cats_out + NN;
    float* keep_out   = scores_out + NN;

    unsigned short* abf = (unsigned short*)d_ws;                       // 4 MB bf16
    int* minHit = (int*)((char*)d_ws + (size_t)NN * DD * sizeof(unsigned short));
    int* cnt    = minHit + NN;

    const dim3 blk(256);

    k_normalize<<<dim3(NN / 8), blk, 0, stream>>>(kern, abf, minHit);

    // block tile 64 rows x 128 cols (8 waves x 32x32); grid covers full square,
    // sub-diagonal wave tiles early-exit uniformly.
    k_sim<<<dim3(NN / 128, NN / 64), blk, 0, stream>>>(abf, cats, minHit);

    k_prep<<<dim3(NN / 8), blk, 0, stream>>>(cats, scores, minHit,
                                             fused, cats_out, scores_out, keep_out, cnt);

    k_aggregate<<<dim3(NN / 8), blk, 0, stream>>>(kern, minHit, cnt, fused);

    k_finalize<<<dim3(NN / 8), blk, 0, stream>>>(minHit, cnt, fused);
}
